// CrossAttention_13761075216695
// MI455X (gfx1250) — matrix operand order repaired and checked by reading, not compiled, not run on hardware
//
#include <hip/hip_runtime.h>
#include <hip/hip_bf16.h>

typedef __attribute__((ext_vector_type(16))) _Float16 v16h;
typedef __attribute__((ext_vector_type(8)))  _Float16 v8h;
typedef __attribute__((ext_vector_type(8)))  float    v8f;
typedef int vi4 __attribute__((vector_size(16)));   // matches builtin param type

#define B_   2
#define N_   4096
#define DIM_ 256
#define H_   8
#define HD_  32
// softmax in log2 domain: fold KVDIM^-0.5 and log2(e) into one scale
#define SL_  (0.0625f * 1.44269504088896f)

// gfx1250 async global->LDS staging (guarded: falls back to load+ds_store)
#if __has_builtin(__builtin_amdgcn_global_load_async_to_lds_b128)
#define ASYNC_STAGE 1
#else
#define ASYNC_STAGE 0
#endif

#if __has_builtin(__builtin_amdgcn_s_wait_asynccnt)
#define ASYNC_WAIT() __builtin_amdgcn_s_wait_asynccnt(0)
#else
#define ASYNC_WAIT() asm volatile("s_wait_asynccnt 0x0" ::: "memory")
#endif

#if ASYNC_STAGE
__device__ __forceinline__ void stage_async(const _Float16* g, _Float16* l) {
  __builtin_amdgcn_global_load_async_to_lds_b128(
      (__attribute__((address_space(1))) vi4*)g,
      (__attribute__((address_space(3))) vi4*)l, 0, 0);
}
#endif

__device__ __forceinline__ v8f wmma_f16(v16h a, v16h b, v8f c) {
  return __builtin_amdgcn_wmma_f32_16x16x32_f16(
      false, a, false, b, (short)0, c, false, false);
}

// ---------------------------------------------------------------------------
// Streaming f32 -> f16 convert (8 elements/thread). n % 8 == 0 for all uses.
// ---------------------------------------------------------------------------
__global__ __launch_bounds__(256) void cvt_kernel(
    const float* __restrict__ src, _Float16* __restrict__ dst, int n)
{
  const int i = (blockIdx.x * 256 + threadIdx.x) * 8;
  if (i + 8 <= n) {
    const float4 a = *(const float4*)(src + i);
    const float4 b = *(const float4*)(src + i + 4);
    v8h o;
    o[0] = (_Float16)a.x; o[1] = (_Float16)a.y;
    o[2] = (_Float16)a.z; o[3] = (_Float16)a.w;
    o[4] = (_Float16)b.x; o[5] = (_Float16)b.y;
    o[6] = (_Float16)b.z; o[7] = (_Float16)b.w;
    *(v8h*)(dst + i) = o;
  }
}

// ---------------------------------------------------------------------------
// Unified f16 GEMM: out[m, o] = sum_k A[m,k] * Wh[o,k] + bias[o]
// A: (8192, 256) f16 row-major; Wh: (256, 256) f16 row-major (out x in).
// Register-blocked 16x64 per wave. Epilogue by mode:
//   0: f16 q/k layout [b][h][n][d]     1: f16 vT layout [b][h][d][n]
//   2: f32 row-major [m][256]
// ---------------------------------------------------------------------------
__global__ __launch_bounds__(256) void gemm16_kernel(
    const _Float16* __restrict__ A, const _Float16* __restrict__ Wh,
    const float* __restrict__ bias, _Float16* __restrict__ out16,
    float* __restrict__ out32, int mode)
{
  const int lane = threadIdx.x & 31;
  const int wave = threadIdx.x >> 5;
  const int tile = blockIdx.x * 8 + wave;      // 2048 wave-tiles
  const int quad   = tile & 3;
  const int tile_m = tile >> 2;                // 0..511
  const int m0 = tile_m * 16;
  const int o0 = quad * 64;
  const int half = lane >> 4;
  const int l15  = lane & 15;

  const _Float16* arow = A + (size_t)(m0 + l15) * DIM_;

  v8f acc[4] = {{}, {}, {}, {}};
  for (int kk = 0; kk < DIM_; kk += 32) {
    v16h a;
    {
      const _Float16* p = arow + kk + half * 16;
      v8h c0 = *(const v8h*)p;
      v8h c1 = *(const v8h*)(p + 8);
      #pragma unroll
      for (int i = 0; i < 8; ++i) { a[i] = c0[i]; a[8 + i] = c1[i]; }
    }
    #pragma unroll
    for (int t = 0; t < 4; ++t) {
      const _Float16* wp =
          Wh + (size_t)(o0 + t * 16 + l15) * DIM_ + kk + half * 16;
      v8h b0 = *(const v8h*)wp;
      v8h b1 = *(const v8h*)(wp + 8);
      v16h b;
      #pragma unroll
      for (int i = 0; i < 8; ++i) { b[i] = b0[i]; b[8 + i] = b1[i]; }
      acc[t] = wmma_f16(a, b, acc[t]);
    }
  }

  #pragma unroll
  for (int t = 0; t < 4; ++t) {
    const int ocol = o0 + t * 16 + l15;
    const float bval = bias[ocol];
    const int hh = ocol >> 5;
    const int dd = ocol & 31;
    if (mode == 0) {
      #pragma unroll
      for (int r = 0; r < 8; ++r) {
        const int m   = m0 + r + half * 8;
        const int bb_ = m >> 12;
        const int n   = m & (N_ - 1);
        out16[(((size_t)(bb_ * H_ + hh) * N_ + n) << 5) + dd] =
            (_Float16)(acc[t][r] + bval);
      }
    } else if (mode == 1) {
      const int bb_   = m0 >> 12;
      const int nbase = (m0 & (N_ - 1)) + half * 8;
      v8h pk;
      #pragma unroll
      for (int r = 0; r < 8; ++r) pk[r] = (_Float16)(acc[t][r] + bval);
      *(v8h*)(out16 + ((size_t)(bb_ * H_ + hh) * HD_ + dd) * N_ + nbase) = pk;
    } else {
      #pragma unroll
      for (int r = 0; r < 8; ++r) {
        const int m = m0 + r + half * 8;
        out32[(size_t)m * DIM_ + ocol] = acc[t][r] + bval;
      }
    }
  }
}

// ---------------------------------------------------------------------------
// Flash attention, cooperative LDS staging (async-to-LDS when available).
// Block = 8 waves, all on the same (b,h); wave w handles q-tile group*8+w.
// Each 32-key step: K-block (32x32 f16) and V-block (32x32 f16) staged into
// double-buffered LDS, then every wave reads its WMMA B-operands from LDS.
// ---------------------------------------------------------------------------
__global__ __launch_bounds__(256) void attn_kernel(
    const _Float16* __restrict__ qh, const _Float16* __restrict__ kh,
    const _Float16* __restrict__ vT, _Float16* __restrict__ attn_out)
{
  __shared__ _Float16 kbuf[2][32 * 32];       // 2 x 2KB
  __shared__ _Float16 vbuf[2][32 * 32];       // 2 x 2KB
  __shared__ _Float16 ptile[8][16 * 32];      // 8 x 1KB P slabs

  const int tid  = threadIdx.x;
  const int lane = tid & 31;
  const int wave = tid >> 5;
  const int half = lane >> 4;
  const int l15  = lane & 15;

  const int bh  = blockIdx.x >> 5;              // 0..15 (b*8+h)
  const int qt  = (blockIdx.x & 31) * 8 + wave; // q tile 0..255
  const int b   = bh >> 3;
  const int h   = bh & 7;

  const _Float16* qb = qh + (size_t)bh * N_ * HD_;
  const _Float16* kb = kh + (size_t)bh * N_ * HD_;   // [n][d]
  const _Float16* vb = vT + (size_t)bh * N_ * HD_;   // [d][n]
  _Float16* lp = ptile[wave];

  // staging role (wave-uniform split): waves 0-3 -> K, waves 4-7 -> V
  const int isV = tid >> 7;
  const int row = (tid & 127) >> 2;           // 0..31
  const int chk = (tid & 3) * 8;              // half-offset 0/8/16/24

  // Q tile as WMMA A operand
  v16h aq;
  {
    const _Float16* p = qb + (size_t)(qt * 16 + l15) * HD_ + half * 16;
    v8h c0 = *(const v8h*)p;
    v8h c1 = *(const v8h*)(p + 8);
    #pragma unroll
    for (int i = 0; i < 8; ++i) { aq[i] = c0[i]; aq[8 + i] = c1[i]; }
  }

  v16h onesv;
  #pragma unroll
  for (int i = 0; i < 16; ++i) onesv[i] = (_Float16)1.0f;

  float mrow[8];
  v8f O0 = {}, O1 = {}, Lacc = {};
  #pragma unroll
  for (int r = 0; r < 8; ++r) mrow[r] = -3.0e38f;

  // ---- preload step 0 into buffer 0 ----
#if ASYNC_STAGE
  {
    const _Float16* g0 = isV ? vb + (size_t)row * N_ + chk
                             : kb + (size_t)row * HD_ + chk;
    _Float16* l0 = isV ? &vbuf[0][row * 32 + chk] : &kbuf[0][row * 32 + chk];
    stage_async(g0, l0);
    ASYNC_WAIT();
  }
#else
  {
    v8h st = isV ? *(const v8h*)(vb + (size_t)row * N_ + chk)
                 : *(const v8h*)(kb + (size_t)row * HD_ + chk);
    if (isV) *(v8h*)(&vbuf[0][row * 32 + chk]) = st;
    else     *(v8h*)(&kbuf[0][row * 32 + chk]) = st;
  }
#endif
  __syncthreads();

  int cur = 0;
  for (int j0 = 0; j0 < N_; j0 += 32) {
    const bool more = (j0 + 32) < N_;     // block-uniform
#if ASYNC_STAGE
    if (more) {   // issue next-block copy early; overlaps with compute below
      const _Float16* gn = isV ? vb + (size_t)row * N_ + (j0 + 32) + chk
                               : kb + (size_t)(j0 + 32 + row) * HD_ + chk;
      _Float16* ln = isV ? &vbuf[cur ^ 1][row * 32 + chk]
                         : &kbuf[cur ^ 1][row * 32 + chk];
      stage_async(gn, ln);
    }
#else
    v8h nst;
    if (more) {
      nst = isV ? *(const v8h*)(vb + (size_t)row * N_ + (j0 + 32) + chk)
                : *(const v8h*)(kb + (size_t)(j0 + 32 + row) * HD_ + chk);
    }
#endif

    // ---- S = Q * K^T from LDS K-block ----
    v16h bk0, bk1;
    {
      const _Float16* p0 = &kbuf[cur][l15 * 32 + half * 16];
      const _Float16* p1 = &kbuf[cur][(16 + l15) * 32 + half * 16];
      v8h x0 = *(const v8h*)p0, x1 = *(const v8h*)(p0 + 8);
      v8h y0 = *(const v8h*)p1, y1 = *(const v8h*)(p1 + 8);
      #pragma unroll
      for (int i = 0; i < 8; ++i) {
        bk0[i] = x0[i]; bk0[8 + i] = x1[i];
        bk1[i] = y0[i]; bk1[8 + i] = y1[i];
      }
    }
    v8f z = {};
    v8f S0 = wmma_f16(aq, bk0, z);
    v8f S1 = wmma_f16(aq, bk1, z);

    // ---- online softmax (log2 domain) ----
    #pragma unroll
    for (int r = 0; r < 8; ++r) {
      float s0 = S0[r] * SL_, s1 = S1[r] * SL_;
      float t = fmaxf(s0, s1);
      t = fmaxf(t, __shfl_xor(t, 1, 32));
      t = fmaxf(t, __shfl_xor(t, 2, 32));
      t = fmaxf(t, __shfl_xor(t, 4, 32));
      t = fmaxf(t, __shfl_xor(t, 8, 32));
      const float mn = fmaxf(mrow[r], t);
      const float cf = exp2f(mrow[r] - mn);
      mrow[r] = mn;
      const float p0 = exp2f(s0 - mn);
      const float p1 = exp2f(s1 - mn);
      O0[r]   *= cf;
      O1[r]   *= cf;
      Lacc[r] *= cf;
      const int mi = r + half * 8;
      lp[mi * 32 + l15]      = (_Float16)p0;
      lp[mi * 32 + 16 + l15] = (_Float16)p1;
    }

    // ---- reload P in A layout (in-wave LDS RAW is ordered) ----
    v16h ap;
    {
      const _Float16* pp = lp + l15 * 32 + half * 16;
      v8h c0 = *(const v8h*)pp;
      v8h c1 = *(const v8h*)(pp + 8);
      #pragma unroll
      for (int i = 0; i < 8; ++i) { ap[i] = c0[i]; ap[8 + i] = c1[i]; }
    }

    // ---- row sums via WMMA against ones ----
    Lacc = wmma_f16(ap, onesv, Lacc);

    // ---- O += P * V from LDS V-block ----
    v16h bv0, bv1;
    {
      const _Float16* p0 = &vbuf[cur][l15 * 32 + half * 16];
      const _Float16* p1 = &vbuf[cur][(16 + l15) * 32 + half * 16];
      v8h x0 = *(const v8h*)p0, x1 = *(const v8h*)(p0 + 8);
      v8h y0 = *(const v8h*)p1, y1 = *(const v8h*)(p1 + 8);
      #pragma unroll
      for (int i = 0; i < 8; ++i) {
        bv0[i] = x0[i]; bv0[8 + i] = x1[i];
        bv1[i] = y0[i]; bv1[8 + i] = y1[i];
      }
    }
    O0 = wmma_f16(ap, bv0, O0);
    O1 = wmma_f16(ap, bv1, O1);

    // ---- settle staging of next block, then block-wide barrier ----
#if ASYNC_STAGE
    if (more) ASYNC_WAIT();
#else
    if (more) {
      if (isV) *(v8h*)(&vbuf[cur ^ 1][row * 32 + chk]) = nst;
      else     *(v8h*)(&kbuf[cur ^ 1][row * 32 + chk]) = nst;
    }
#endif
    __syncthreads();
    cur ^= 1;
  }

  // ---- normalize and store merged-head f16 output [b][n][256] ----
  #pragma unroll
  for (int r = 0; r < 8; ++r) {
    const float inv = 1.0f / Lacc[r];
    const int n = qt * 16 + r + half * 8;
    const size_t base = ((size_t)b * N_ + n) * DIM_ + h * HD_;
    attn_out[base + l15]      = (_Float16)(O0[r] * inv);
    attn_out[base + 16 + l15] = (_Float16)(O1[r] * inv);
  }
}

// ---------------------------------------------------------------------------
extern "C" void kernel_launch(void* const* d_in, const int* in_sizes, int n_in,
                              void* d_out, int out_size, void* d_ws, size_t ws_size,
                              hipStream_t stream)
{
  const float* x  = (const float*)d_in[0];
  const float* y  = (const float*)d_in[1];
  const float* Wq = (const float*)d_in[2];
  const float* bq = (const float*)d_in[3];
  const float* Wk = (const float*)d_in[4];
  const float* bk = (const float*)d_in[5];
  const float* Wv = (const float*)d_in[6];
  const float* bv = (const float*)d_in[7];
  const float* Wo = (const float*)d_in[8];
  const float* bo = (const float*)d_in[9];
  float* out = (float*)d_out;

  _Float16* ws = (_Float16*)d_ws;
  const size_t per  = (size_t)B_ * H_ * N_ * HD_;  // 2M halves
  const size_t wsz  = (size_t)DIM_ * DIM_;         // 64K halves
  _Float16* qh   = ws;
  _Float16* kh   = ws + per;
  _Float16* vTt  = ws + 2 * per;
  _Float16* ao   = ws + 3 * per;
  _Float16* xh   = ws + 4 * per;
  _Float16* yh   = ws + 5 * per;
  _Float16* wq16 = ws + 6 * per;
  _Float16* wk16 = wq16 + wsz;
  _Float16* wv16 = wk16 + wsz;
  _Float16* wo16 = wv16 + wsz;

  dim3 blk(256);
  const int nx = B_ * N_ * DIM_;                   // 2M elements
  const int nw = DIM_ * DIM_;                      // 64K elements

  // f32 -> f16 pre-conversion (removes all cvt work from GEMM hot loops)
  cvt_kernel<<<nx / (256 * 8), blk, 0, stream>>>(x, xh, nx);
  cvt_kernel<<<nx / (256 * 8), blk, 0, stream>>>(y, yh, nx);
  cvt_kernel<<<nw / (256 * 8), blk, 0, stream>>>(Wq, wq16, nw);
  cvt_kernel<<<nw / (256 * 8), blk, 0, stream>>>(Wk, wk16, nw);
  cvt_kernel<<<nw / (256 * 8), blk, 0, stream>>>(Wv, wv16, nw);
  cvt_kernel<<<nw / (256 * 8), blk, 0, stream>>>(Wo, wo16, nw);

  // QKV projections (2048 wave-tiles -> 256 blocks)
  gemm16_kernel<<<256, blk, 0, stream>>>(xh, wq16, bq, qh,  nullptr, 0);
  gemm16_kernel<<<256, blk, 0, stream>>>(yh, wk16, bk, kh,  nullptr, 0);
  gemm16_kernel<<<256, blk, 0, stream>>>(yh, wv16, bv, vTt, nullptr, 1);

  // attention: 512 blocks x 8 waves (8 q-tiles of one head per block)
  attn_kernel<<<512, blk, 0, stream>>>(qh, kh, vTt, ao);

  // output projection -> f32
  gemm16_kernel<<<256, blk, 0, stream>>>(ao, wo16, bo, nullptr, out, 2);
}
